// NeuralCDE_15341623181673
// MI455X (gfx1250) — matrix operand order repaired and checked by reading, not compiled, not run on hardware
//
#include <hip/hip_runtime.h>
#include <hip/hip_bf16.h>

typedef __attribute__((ext_vector_type(16))) _Float16 v16h;
typedef __attribute__((ext_vector_type(8)))  _Float16 v8h;
typedef __attribute__((ext_vector_type(8)))  float    v8f;

#define B_TOT   4096
#define LSEG    511          // L-1 segments
#define C_IN    8
#define H_DIM   64
#define HHD     128
#define MT      16           // batch rows per workgroup
#define NTHREADS 128         // 4 waves: fatter per-wave work between barriers

// fragment-region offsets in d_ws (in f16 elements)
#define WIN_OFF   0          // W_in : 2 kb x 8 nb  = 16 frags  * 512 = 8192
#define WH_OFF    8192       // W_h  : 2 layers * (4 kb x 8 nb) = 64 frags = 32768
#define WOUT_OFF  40960      // W_out: 4 kb x 32 nb = 128 frags = 65536
#define W_TOTAL   106496

// padded LDS strides (f16 elements, keep 16B alignment of sub-rows)
#define ZIN_STR 72
#define H_STR   136

// ---------------------------------------------------------------------------
// Prep: convert fp32 weights to f16, pre-swizzled into WMMA B-fragment order.
// Fragment (kb,nb) = K-block of 32, N-block of 16. Within a fragment:
//   element index = lane*16 + i, value = W[kb*32 + (lane>>4)*16 + i][nb*16 + (lane&15)]
// ---------------------------------------------------------------------------
__global__ void prep_weights_kernel(const float* __restrict__ W_in,
                                    const float* __restrict__ W_h,
                                    const float* __restrict__ W_out,
                                    _Float16* __restrict__ wsf) {
    int tid = blockIdx.x * blockDim.x + threadIdx.x;
    if (tid >= W_TOTAL) return;
    float v;
    if (tid < WH_OFF) {
        int f = tid >> 9, rem = tid & 511;
        int kb = f >> 3, nb = f & 7;
        int lane = rem >> 4, i = rem & 15;
        int k = kb * 32 + ((lane >> 4) << 4) + i;
        int n = nb * 16 + (lane & 15);
        v = W_in[k * HHD + n];
    } else if (tid < WOUT_OFF) {
        int t2 = tid - WH_OFF;
        int layer = t2 >> 14, r2 = t2 & 16383;
        int f = r2 >> 9, rem = r2 & 511;
        int kb = f >> 3, nb = f & 7;
        int lane = rem >> 4, i = rem & 15;
        int k = kb * 32 + ((lane >> 4) << 4) + i;
        int n = nb * 16 + (lane & 15);
        v = W_h[layer * HHD * HHD + k * HHD + n];
    } else {
        int t3 = tid - WOUT_OFF;
        int f = t3 >> 9, rem = t3 & 511;
        int kb = f >> 5, nb = f & 31;
        int lane = rem >> 4, i = rem & 15;
        int k = kb * 32 + ((lane >> 4) << 4) + i;
        int n = nb * 16 + (lane & 15);
        v = W_out[k * 512 + n];
    }
    wsf[tid] = (_Float16)v;
}

// ---------------------------------------------------------------------------
// A fragment: lane l holds row r=l&15; K elems {kb*32+kh*16 .. +7} and
// {kb*32+kh*16+8 .. +7} with kh = l>>4. Two contiguous 16B LDS chunks.
// ---------------------------------------------------------------------------
__device__ __forceinline__ v16h load_a_frag(const _Float16* h, int r, int kh,
                                            int kb, int stride) {
    const v8h* lo = (const v8h*)(h + r * stride + kb * 32 + kh * 16);
    const v8h* hi = (const v8h*)(h + r * stride + kb * 32 + kh * 16 + 8);
    v8h a0 = *lo, a1 = *hi;
    v16h a;
#pragma unroll
    for (int i = 0; i < 8; i++) { a[i] = a0[i]; a[8 + i] = a1[i]; }
    return a;
}

__device__ __forceinline__ v16h load_b_frag(const _Float16* w) {
    const v8h* p = (const v8h*)w;
    v8h b0 = p[0], b1 = p[1];
    v16h b;
#pragma unroll
    for (int i = 0; i < 8; i++) { b[i] = b0[i]; b[8 + i] = b1[i]; }
    return b;
}

// One dense layer: 16xK (LDS f16) @ Kx16N (ws f16) + bias, relu -> LDS f16.
// Each of 4 waves computes 2 adjacent N-tiles; A-fragments loaded once.
template <int NKB>
__device__ __forceinline__ void dense_relu2(const _Float16* src, int srcStride,
                                            const _Float16* wbase, int fragPitch,
                                            const float* __restrict__ bias,
                                            _Float16* dst, int dstStride,
                                            int lane, int wave) {
    int r = lane & 15, kh = (lane >> 4) & 1;
    v16h a[NKB];
#pragma unroll
    for (int kb = 0; kb < NKB; kb++) a[kb] = load_a_frag(src, r, kh, kb, srcStride);
#pragma unroll
    for (int jn = 0; jn < 2; jn++) {
        int nb = wave * 2 + jn;
        v8f acc = {};
#pragma unroll
        for (int kb = 0; kb < NKB; kb++) {
            v16h b = load_b_frag(wbase + (size_t)(kb * fragPitch + nb) * 512 + lane * 16);
            acc = __builtin_amdgcn_wmma_f32_16x16x32_f16(
                false, a[kb], false, b, (short)0, acc, false, false);
        }
        int n = nb * 16 + r;
        float bs = bias[n];
#pragma unroll
        for (int v = 0; v < 8; v++) {
            float x = acc[v] + bs;
            x = x > 0.f ? x : 0.f;
            dst[(v + kh * 8) * dstStride + n] = (_Float16)x;
        }
    }
}

__device__ __forceinline__ float fast_tanh(float x) {
    return 1.f - 2.f / (__expf(2.f * x) + 1.f);
}

// ---------------------------------------------------------------------------
// Main persistent kernel: one WG = 16 batch rows, 4 waves, 511 RK4 steps.
// ---------------------------------------------------------------------------
__global__ __launch_bounds__(NTHREADS)
void cde_rk4_kernel(const float* __restrict__ coeffs,
                    const float* __restrict__ W_init,
                    const float* __restrict__ b_init,
                    const float* __restrict__ b_in,
                    const float* __restrict__ b_h,
                    const float* __restrict__ b_out,
                    const float* __restrict__ W_read,
                    const float* __restrict__ b_read,
                    const _Float16* __restrict__ wsf,
                    float* __restrict__ out) {
    __shared__ _Float16 sZin[MT * ZIN_STR];
    __shared__ _Float16 sHA[MT * H_STR];
    __shared__ _Float16 sHB[MT * H_STR];
    __shared__ float    sZ[MT * H_DIM];
    __shared__ float    sK[4][MT * H_DIM];
    __shared__ float    sDX[MT * C_IN];

    const int tid = threadIdx.x;
    const int lane = tid & 31, wave = tid >> 5;
    const int bbase = blockIdx.x * MT;

    // --- z0 = a[:,0,:] @ W_init + b_init  (tiny K=8 GEMM, scalar) ---
#pragma unroll
    for (int q = 0; q < 8; q++) {
        int idx = tid + q * NTHREADS;          // 0..1023
        int r = idx >> 6, hcol = idx & 63;
        const float* arow = coeffs + (size_t)(bbase + r) * LSEG * 32;
        float acc = b_init[hcol];
#pragma unroll
        for (int c = 0; c < C_IN; c++) acc += arow[c] * W_init[c * H_DIM + hcol];
        sZ[idx] = acc;
    }
    __syncthreads();

    // software-pipelined segment coefficients (one (row,chan) per thread; 128 = 16x8)
    const int rr = tid >> 3, cc = tid & 7;
    float bn, cn, dn, bnN, cnN, dnN;
    {
        size_t base = ((size_t)(bbase + rr) * LSEG) * 32;
        bnN = coeffs[base + 8 + cc];
        cnN = coeffs[base + 16 + cc];
        dnN = coeffs[base + 24 + cc];
    }

    for (int t = 0; t < LSEG; t++) {
        bn = bnN; cn = cnN; dn = dnN;
        if (t + 1 < LSEG) {                     // prefetch next step's coeffs
            size_t base = ((size_t)(bbase + rr) * LSEG + (t + 1)) * 32;
            bnN = coeffs[base + 8 + cc];
            cnN = coeffs[base + 16 + cc];
            dnN = coeffs[base + 24 + cc];
        }
        for (int e = 0; e < 4; e++) {
            // dX(frac) for this RK stage (every thread owns one (row,chan))
            {
                float frac = (e == 0) ? 0.f : (e == 1) ? (1.f / 3.f)
                             : (e == 2) ? (2.f / 3.f) : 1.f;
                sDX[tid] = bn + (cn + dn * frac) * frac;
            }
            // zin per Kutta 3/8 rule -> f16 staging
#pragma unroll
            for (int q = 0; q < 8; q++) {
                int idx = tid + q * NTHREADS;
                float val;
                if (e == 0)      val = sZ[idx];
                else if (e == 1) val = sZ[idx] + sK[0][idx] * (1.f / 3.f);
                else if (e == 2) val = sZ[idx] + sK[1][idx] - sK[0][idx] * (1.f / 3.f);
                else             val = sZ[idx] + sK[0][idx] - sK[1][idx] + sK[2][idx];
                int r = idx >> 6, col = idx & 63;
                sZin[r * ZIN_STR + col] = (_Float16)val;
            }
            __syncthreads();

            // MLP: 64 -> 128 -> 128 -> 128 (relu); 2 N-tiles per wave per layer
            dense_relu2<2>(sZin, ZIN_STR, wsf + WIN_OFF, 8, b_in, sHA, H_STR, lane, wave);
            __syncthreads();
            dense_relu2<4>(sHA, H_STR, wsf + WH_OFF, 8, b_h, sHB, H_STR, lane, wave);
            __syncthreads();
            dense_relu2<4>(sHB, H_STR, wsf + WH_OFF + 16384, 8, b_h + HHD, sHA, H_STR, lane, wave);
            __syncthreads();

            // 128 -> 512, tanh, contract over C=8 with dX, write k_e
            // 8 N-tiles per wave; A-fragments loaded once, reused across tiles
            {
                int r = lane & 15, kh = (lane >> 4) & 1;
                v16h afr[4];
#pragma unroll
                for (int kb = 0; kb < 4; kb++) afr[kb] = load_a_frag(sHA, r, kh, kb, H_STR);
#pragma unroll
                for (int j = 0; j < 8; j++) {
                    int nb = wave * 8 + j;
                    v8f acc = {};
#pragma unroll
                    for (int kb = 0; kb < 4; kb++) {
                        v16h b = load_b_frag(wsf + WOUT_OFF +
                                             (size_t)(kb * 32 + nb) * 512 + lane * 16);
                        acc = __builtin_amdgcn_wmma_f32_16x16x32_f16(
                            false, afr[kb], false, b, (short)0, acc, false, false);
                    }
                    int gcol = nb * 16 + r;
                    float bs = b_out[gcol];
                    int c = gcol & 7, h = gcol >> 3;
                    float vals[8];
#pragma unroll
                    for (int v = 0; v < 8; v++) {
                        float x = fast_tanh(acc[v] + bs);
                        int row = v + kh * 8;
                        vals[v] = x * sDX[row * 8 + c];
                    }
                    // butterfly-reduce the 8 channel lanes (lanes sharing l>>3)
#pragma unroll
                    for (int m = 1; m < 8; m <<= 1) {
#pragma unroll
                        for (int v = 0; v < 8; v++)
                            vals[v] += __shfl_xor(vals[v], m, 32);
                    }
                    if ((lane & 7) == 0) {
#pragma unroll
                        for (int v = 0; v < 8; v++)
                            sK[e][(v + kh * 8) * H_DIM + h] = vals[v];
                    }
                }
            }
            __syncthreads();
        }
        // z += (k1 + 3(k2+k3) + k4)/8
#pragma unroll
        for (int q = 0; q < 8; q++) {
            int idx = tid + q * NTHREADS;
            sZ[idx] += (sK[0][idx] + 3.f * (sK[1][idx] + sK[2][idx]) + sK[3][idx]) * 0.125f;
        }
        __syncthreads();
    }

    // readout: out = z @ W_read + b_read   (O=1)
    if (tid < MT) {
        float acc = b_read[0];
        for (int h = 0; h < H_DIM; h++) acc += sZ[tid * H_DIM + h] * W_read[h];
        out[bbase + tid] = acc;
    }
}

extern "C" void kernel_launch(void* const* d_in, const int* in_sizes, int n_in,
                              void* d_out, int out_size, void* d_ws, size_t ws_size,
                              hipStream_t stream) {
    const float* coeffs = (const float*)d_in[0];
    const float* W_init = (const float*)d_in[1];
    const float* b_init = (const float*)d_in[2];
    const float* W_in   = (const float*)d_in[3];
    const float* b_in   = (const float*)d_in[4];
    const float* W_h    = (const float*)d_in[5];
    const float* b_h    = (const float*)d_in[6];
    const float* W_out  = (const float*)d_in[7];
    const float* b_out  = (const float*)d_in[8];
    const float* W_read = (const float*)d_in[9];
    const float* b_read = (const float*)d_in[10];
    float* out = (float*)d_out;
    _Float16* wsf = (_Float16*)d_ws;

    // 1) swizzle weights to f16 WMMA B-fragment layout in workspace
    prep_weights_kernel<<<(W_TOTAL + 255) / 256, 256, 0, stream>>>(
        W_in, W_h, W_out, wsf);

    // 2) persistent RK4 integration: 256 WGs x 16 batch rows x 4 waves
    cde_rk4_kernel<<<B_TOT / MT, NTHREADS, 0, stream>>>(
        coeffs, W_init, b_init, b_in, b_h, b_out, W_read, b_read, wsf, out);
}
